// StockEmbeddingAttention_11725260718397
// MI455X (gfx1250) — matrix operand order repaired and checked by reading, not compiled, not run on hardware
//
#include <hip/hip_runtime.h>
#include <hip/hip_bf16.h>

typedef __attribute__((ext_vector_type(16))) _Float16 v16h;
typedef __attribute__((ext_vector_type(8)))  float    v8f;

constexpr int kB = 128;    // batch
constexpr int kN = 2000;   // sequence length
constexpr int kF = 128;    // feature dim
constexpr int kChunks = 16;              // chunks per batch for the streaming pass
constexpr int kPartStride = 132;         // m, s, w[128], pad to keep 16B align

// ---------------------------------------------------------------------------
// Kernel 1: v[b,:] = (x_last[b] @ Wk + bk) @ Wk^T  via two 128x128x128 WMMA GEMMs
// One workgroup, 8 waves; wave w owns output tile-row w (M = 16w..16w+15).
// f16 inputs, f32 accumulate (values are O(1), f16 is plenty).
// ---------------------------------------------------------------------------
__global__ __launch_bounds__(256)
void sea_prep_v(const float* __restrict__ seq, const float* __restrict__ Wk,
                const float* __restrict__ bk, float* __restrict__ v_out)
{
    __shared__ _Float16 Wh[kF * kF];   // 32 KB
    __shared__ _Float16 Ph[kF * kF];   // 32 KB (p_last, f16)

    const int tid  = threadIdx.x;
    const int wv   = tid >> 5;         // wave id 0..7
    const int lane = tid & 31;
    const int rA   = lane & 15;        // row-in-tile for A frag / N-in-tile for B,C,D
    const int hi   = lane >> 4;        // which K-half this lane holds

    for (int i = tid; i < kF * kF; i += 256) Wh[i] = (_Float16)Wk[i];
    __syncthreads();

    const int m0 = wv * 16;

    // ---- GEMM1: P = X_last @ Wk + bk ------------------------------------
    // Preload A fragments (invariant across output tile column j).
    v16h aF[4];
    {
        const int brow = m0 + rA;                       // batch row
        const float* xl = seq + ((size_t)brow * kN + (kN - 1)) * kF;
        for (int kk = 0; kk < 4; ++kk) {
            #pragma unroll
            for (int i = 0; i < 16; ++i) {
                int k = kk * 32 + hi * 16 + i;
                aF[kk][i] = (_Float16)xl[k];
            }
        }
    }
    for (int j = 0; j < 8; ++j) {
        v8f acc = {};
        for (int kk = 0; kk < 4; ++kk) {
            v16h bF;
            #pragma unroll
            for (int r = 0; r < 8; ++r) {
                int k = kk * 32 + hi * 16 + 2 * r;
                int n = 16 * j + rA;
                bF[2 * r]     = Wh[(k)     * kF + n];
                bF[2 * r + 1] = Wh[(k + 1) * kF + n];
            }
            acc = __builtin_amdgcn_wmma_f32_16x16x32_f16(
                false, aF[kk], false, bF, (short)0, acc, false, false);
        }
        const float bkv = bk[16 * j + rA];
        #pragma unroll
        for (int i = 0; i < 8; ++i) {
            int row = m0 + i + hi * 8;
            int col = 16 * j + rA;
            Ph[row * kF + col] = (_Float16)(acc[i] + bkv);
        }
    }
    __syncthreads();

    // ---- GEMM2: V = P @ Wk^T   (B[k][n] = Wk[n][k]) ---------------------
    v16h aP[4];
    for (int kk = 0; kk < 4; ++kk) {
        #pragma unroll
        for (int i = 0; i < 16; ++i) {
            int k = kk * 32 + hi * 16 + i;
            aP[kk][i] = Ph[(m0 + rA) * kF + k];
        }
    }
    for (int j = 0; j < 8; ++j) {
        v8f acc = {};
        for (int kk = 0; kk < 4; ++kk) {
            v16h bF;
            #pragma unroll
            for (int r = 0; r < 8; ++r) {
                int k = kk * 32 + hi * 16 + 2 * r;
                int n = 16 * j + rA;
                bF[2 * r]     = Wh[n * kF + (k)];
                bF[2 * r + 1] = Wh[n * kF + (k + 1)];
            }
            acc = __builtin_amdgcn_wmma_f32_16x16x32_f16(
                false, aP[kk], false, bF, (short)0, acc, false, false);
        }
        #pragma unroll
        for (int i = 0; i < 8; ++i) {
            int row = m0 + i + hi * 8;
            int col = 16 * j + rA;
            v_out[row * kF + col] = acc[i];
        }
    }
}

// ---------------------------------------------------------------------------
// Kernel 2: single streaming pass over sequence_features with online softmax.
// Grid (chunk, b); one 512B row per wave per iteration, perfectly coalesced.
// Emits per-(b,chunk) partial {m, s, w[128]} to workspace.
// ---------------------------------------------------------------------------
__global__ __launch_bounds__(256)
void sea_stream_pass(const float* __restrict__ seq, const float* __restrict__ v_ws,
                     float* __restrict__ partials)
{
    const int b     = blockIdx.y;
    const int chunk = blockIdx.x;
    const int tid   = threadIdx.x;
    const int wv    = tid >> 5;
    const int lane  = tid & 31;

    constexpr int rowsPer = (kN + kChunks - 1) / kChunks;   // 125
    const int nBeg = chunk * rowsPer;
    const int nEndRaw = nBeg + rowsPer;
    const int nEnd = (nEndRaw < kN) ? nEndRaw : kN;

    const float4 vl = ((const float4*)(v_ws + b * kF))[lane];
    const float invs = 0.08838834764831843f;  // 1/sqrt(128)

    float  m = -3.0e38f, ssum = 0.f;
    float4 w = make_float4(0.f, 0.f, 0.f, 0.f);

    for (int n = nBeg + wv; n < nEnd; n += 8) {
        const float* rowp = seq + ((size_t)b * kN + n) * kF;
        __builtin_prefetch((const void*)(rowp + 8 * kF), 0, 0);  // speculative, OOB-safe
        float4 xv = ((const float4*)rowp)[lane];
        float d = xv.x * vl.x + xv.y * vl.y + xv.z * vl.z + xv.w * vl.w;
        #pragma unroll
        for (int off = 16; off > 0; off >>= 1) d += __shfl_xor(d, off, 32);
        float s = d * invs;

        float mn    = fmaxf(m, s);
        float alpha = __expf(m - mn);
        float e     = __expf(s - mn);
        ssum = ssum * alpha + e;
        w.x = w.x * alpha + e * xv.x;
        w.y = w.y * alpha + e * xv.y;
        w.z = w.z * alpha + e * xv.z;
        w.w = w.w * alpha + e * xv.w;
        m = mn;
    }

    __shared__ float sm[8], ss[8];
    __shared__ float sw[8][kF];
    if (lane == 0) { sm[wv] = m; ss[wv] = ssum; }
    ((float4*)sw[wv])[lane] = w;
    __syncthreads();

    if (tid < kF) {
        float mg = sm[0];
        #pragma unroll
        for (int i = 1; i < 8; ++i) mg = fmaxf(mg, sm[i]);
        float wt = 0.f, st = 0.f;
        #pragma unroll
        for (int i = 0; i < 8; ++i) {
            float al = __expf(sm[i] - mg);
            wt += sw[i][tid] * al;
            st += ss[i] * al;
        }
        float* pp = partials + (size_t)(b * kChunks + chunk) * kPartStride;
        pp[2 + tid] = wt;
        if (tid == 0) { pp[0] = mg; pp[1] = st; }
    }
}

// ---------------------------------------------------------------------------
// Kernel 3: combine chunk partials, append x_last.
// out = [weighted, last] ; stage-2 softmax sums to 1, and 1+1e-8 == 1 in f32.
// ---------------------------------------------------------------------------
__global__ __launch_bounds__(128)
void sea_finalize(const float* __restrict__ seq, const float* __restrict__ partials,
                  float* __restrict__ out)
{
    const int b = blockIdx.x;
    const int f = threadIdx.x;
    const float* pb = partials + (size_t)b * kChunks * kPartStride;

    float mg = pb[0];
    for (int c = 1; c < kChunks; ++c) mg = fmaxf(mg, pb[c * kPartStride]);
    float st = 0.f, wt = 0.f;
    for (int c = 0; c < kChunks; ++c) {
        float al = __expf(pb[c * kPartStride] - mg);
        st += pb[c * kPartStride + 1] * al;
        wt += pb[c * kPartStride + 2 + f] * al;
    }
    const float denom = 1.0f + 1e-8f;   // == 1.0f in f32, kept for fidelity
    out[(size_t)b * (2 * kF) + f]      = (wt / st) / denom;
    out[(size_t)b * (2 * kF) + kF + f] =
        seq[((size_t)b * kN + (kN - 1)) * kF + f] / denom;
}

extern "C" void kernel_launch(void* const* d_in, const int* in_sizes, int n_in,
                              void* d_out, int out_size, void* d_ws, size_t ws_size,
                              hipStream_t stream)
{
    const float* seq = (const float*)d_in[0];
    // d_in[1] stock_ids, d_in[4] emb, d_in[5] attn_w, d_in[6] attn_b:
    // only used through softmax(a2) whose sum is 1 -> algebraically eliminated.
    const float* Wk  = (const float*)d_in[2];
    const float* bk  = (const float*)d_in[3];

    float* ws       = (float*)d_ws;
    float* v_ws     = ws;                 // 128*128 f32 = 64 KB
    float* partials = ws + kF * kF;       // B * kChunks * 132 f32 ≈ 1.08 MB

    sea_prep_v<<<1, 256, 0, stream>>>(seq, Wk, bk, v_ws);
    sea_stream_pass<<<dim3(kChunks, kB), 256, 0, stream>>>(seq, v_ws, partials);
    sea_finalize<<<kB, 128, 0, stream>>>(seq, partials, (float*)d_out);
}
